// CTAttention_19756849562109
// MI455X (gfx1250) — matrix operand order repaired and checked by reading, not compiled, not run on hardware
//
#include <hip/hip_runtime.h>

#define B_ 8
#define L_ 1024
#define H_ 8
#define E_ 64
#define S_ 4
#define BH_ (B_*H_)
#define DQK_ (S_*E_)   // 256 joint (s,e) contraction dim

typedef __attribute__((ext_vector_type(16))) __bf16 v16bf;
typedef __attribute__((ext_vector_type(8)))  float  v8f;

union Frag16 {
    v16bf v;
    unsigned int u[8];
    unsigned short s[16];
};

// A-matrix (16x32 bf16) per-lane k offset of pair p (contiguous, matches B):
// lanes 0-15 = rows, half = lane>>4 selects k 16*half..16*half+15; pairs packed 2/VGPR.
__device__ __forceinline__ int kA0(int p, int half) {
    return 2*p + half*16;
}

__device__ __forceinline__ float redmax16(float x) {
    #pragma unroll
    for (int m = 1; m < 16; m <<= 1) x = fmaxf(x, __shfl_xor(x, m, 32));
    return x;
}
__device__ __forceinline__ float redsum16(float x) {
    #pragma unroll
    for (int m = 1; m < 16; m <<= 1) x += __shfl_xor(x, m, 32);
    return x;
}

// ---- async global->LDS (CDNA5 GLOBAL_LOAD_ASYNC_TO_LDS_B128, ASYNCcnt) ----
// Inline asm (VFLAT encoding: vdst = LDS byte-offset VGPR, vaddr = 64-bit
// global address pair, off = no SADDR). Bypasses the builtin's LangAS pointer
// types which are not spellable from HIP source.
#if defined(__gfx1250__)
#define HAVE_ASYNC 1
__device__ __forceinline__ void async_cp16(void* lds, const void* glob) {
    unsigned lofs = (unsigned)(unsigned long long)lds;   // flat LDS addr low 32b
    unsigned long long g = (unsigned long long)glob;
    asm volatile("global_load_async_to_lds_b128 %0, %1, off"
                 :: "v"(lofs), "v"(g) : "memory");
}
__device__ __forceinline__ void wait_async0() {
#if __has_builtin(__builtin_amdgcn_s_wait_asynccnt)
    __builtin_amdgcn_s_wait_asynccnt(0);
#else
    asm volatile("s_wait_asynccnt 0x0" ::: "memory");
#endif
}
#else
#define HAVE_ASYNC 0
__device__ __forceinline__ void wait_async0() {}
#endif

// ---------------------------------------------------------------------------
// Kernel 1: ct = ((1-tau)*x_l + tau*x_{l+1}) @ W^T + b   (rows = b,l,h,s)
// out bf16 [BH, L, S*E] laid out ((b*H+h)*L + l)*256 + s*64 + f
// ---------------------------------------------------------------------------
__global__ __launch_bounds__(256)
void interp_proj_qk(const float* __restrict__ x, const float* __restrict__ tau,
                    const float* __restrict__ W, const float* __restrict__ bias,
                    __bf16* __restrict__ out) {
    __shared__ __bf16 Wl[E_*E_];
    __shared__ float  bl[E_];
    for (int i = threadIdx.x; i < E_*E_; i += 256) Wl[i] = (__bf16)W[i];
    if (threadIdx.x < E_) bl[threadIdx.x] = bias[threadIdx.x];
    __syncthreads();

    const int wave = threadIdx.x >> 5;
    const int lane = threadIdx.x & 31;
    const int nn   = lane & 15;
    const int half = lane >> 4;
    const int rowbase = blockIdx.x * 128 + wave * 16;

    // --- build A frags: row r = rowbase + (lane&15), k in two 32-chunks
    const int r = rowbase + nn;
    const int s = r & 3, h = (r >> 2) & 7, l = (r >> 5) & 1023, b = r >> 15;
    const int lp = (l < L_-1) ? (l+1) : l;
    const float* xl = x + (((b*L_ + l )*H_ + h) * E_);
    const float* xn = x + (((b*L_ + lp)*H_ + h) * E_);
    const float t = tau[(b*L_ + l)*S_ + s];

    Frag16 a[2];
    #pragma unroll
    for (int kc = 0; kc < 2; ++kc) {
        #pragma unroll
        for (int p = 0; p < 8; ++p) {
            int k0 = kc*32 + kA0(p, half);
            a[kc].v[2*p]   = (__bf16)((1.f - t)*xl[k0]   + t*xn[k0]);
            a[kc].v[2*p+1] = (__bf16)((1.f - t)*xl[k0+1] + t*xn[k0+1]);
        }
    }

    #pragma unroll
    for (int nt = 0; nt < 4; ++nt) {
        v8f c = {0.f,0.f,0.f,0.f,0.f,0.f,0.f,0.f};
        #pragma unroll
        for (int kc = 0; kc < 2; ++kc) {
            Frag16 bf;
            #pragma unroll
            for (int p = 0; p < 8; ++p) {
                // B[k][n] = W[n][k]; lanes0-15 hold K=0..15 of each chunk
                int k0 = kc*32 + half*16 + 2*p;
                bf.u[p] = *(const unsigned*)&Wl[(nt*16 + nn)*E_ + k0];
            }
            c = __builtin_amdgcn_wmma_f32_16x16x32_bf16(
                    false, a[kc].v, false, bf.v, (short)0, c, false, false);
        }
        #pragma unroll
        for (int i = 0; i < 8; ++i) {
            int r2 = rowbase + half*8 + i;   // C/D row = vgpr + 8*half
            int s2 = r2 & 3, h2 = (r2 >> 2) & 7, l2 = (r2 >> 5) & 1023, b2 = r2 >> 15;
            int col = nt*16 + nn;
            out[((b2*H_ + h2)*L_ + l2)*DQK_ + s2*E_ + col] = (__bf16)(c[i] + bl[col]);
        }
    }
}

// ---------------------------------------------------------------------------
// Kernel 2: v_bar = 0.5*((S-T)*x_l + T*x_{l+1}) @ Wv^T + 2*bv,  T = sum_s tau
// out bf16 [BH, L, 64]
// ---------------------------------------------------------------------------
__global__ __launch_bounds__(256)
void interp_vbar(const float* __restrict__ x, const float* __restrict__ tau,
                 const float* __restrict__ W, const float* __restrict__ bias,
                 __bf16* __restrict__ out) {
    __shared__ __bf16 Wl[E_*E_];
    __shared__ float  bl[E_];
    for (int i = threadIdx.x; i < E_*E_; i += 256) Wl[i] = (__bf16)W[i];
    if (threadIdx.x < E_) bl[threadIdx.x] = bias[threadIdx.x];
    __syncthreads();

    const int wave = threadIdx.x >> 5;
    const int lane = threadIdx.x & 31;
    const int nn   = lane & 15;
    const int half = lane >> 4;
    const int rowbase = blockIdx.x * 128 + wave * 16;

    const int r = rowbase + nn;                       // r = (b*L + l)*H + h
    const int h = r & 7, l = (r >> 3) & 1023, b = r >> 13;
    const int lp = (l < L_-1) ? (l+1) : l;
    const float* xl = x + (((b*L_ + l )*H_ + h) * E_);
    const float* xn = x + (((b*L_ + lp)*H_ + h) * E_);
    const float* tp = tau + (b*L_ + l)*S_;
    const float T = tp[0] + tp[1] + tp[2] + tp[3];

    Frag16 a[2];
    #pragma unroll
    for (int kc = 0; kc < 2; ++kc) {
        #pragma unroll
        for (int p = 0; p < 8; ++p) {
            int k0 = kc*32 + kA0(p, half);
            a[kc].v[2*p]   = (__bf16)(0.5f*((4.f - T)*xl[k0]   + T*xn[k0]));
            a[kc].v[2*p+1] = (__bf16)(0.5f*((4.f - T)*xl[k0+1] + T*xn[k0+1]));
        }
    }

    #pragma unroll
    for (int nt = 0; nt < 4; ++nt) {
        v8f c = {0.f,0.f,0.f,0.f,0.f,0.f,0.f,0.f};
        #pragma unroll
        for (int kc = 0; kc < 2; ++kc) {
            Frag16 bf;
            #pragma unroll
            for (int p = 0; p < 8; ++p) {
                int k0 = kc*32 + half*16 + 2*p;
                bf.u[p] = *(const unsigned*)&Wl[(nt*16 + nn)*E_ + k0];
            }
            c = __builtin_amdgcn_wmma_f32_16x16x32_bf16(
                    false, a[kc].v, false, bf.v, (short)0, c, false, false);
        }
        #pragma unroll
        for (int i = 0; i < 8; ++i) {
            int r2 = rowbase + half*8 + i;
            int h2 = r2 & 7, l2 = (r2 >> 3) & 1023, b2 = r2 >> 13;
            int col = nt*16 + nn;
            out[((b2*H_ + h2)*L_ + l2)*E_ + col] = (__bf16)(c[i] + 2.f*bl[col]);
        }
    }
}

// ---------------------------------------------------------------------------
// Kernel 3: causal flash attention, double-buffered async K/V staging.
// Q,K: bf16 [BH, L, 256]; Vb: bf16 [BH, L, 64]; out f32 [B, L, H, 64].
// 256 threads = 8 waves; wave w owns Q rows [qbase + 16w, +16); KV tile = 32.
// Combined softmax scale = 0.5 * (1/sqrt(E)) = 1/16 (0.5 of v_bar prefolded).
// ---------------------------------------------------------------------------
__global__ __launch_bounds__(256)
void attn_fwd(const __bf16* __restrict__ ctq, const __bf16* __restrict__ ctk,
              const __bf16* __restrict__ vbar, float* __restrict__ out) {
    __shared__ __bf16 Kbuf[2][32 * DQK_];  // 2 x 16 KB
    __shared__ __bf16 Vbuf[2][32 * E_];    // 2 x  4 KB
    __shared__ __bf16 Pbuf[8][16 * 32];    //      8 KB (per-wave P transpose)

    const int bh    = blockIdx.y;
    const int qbase = blockIdx.x * 128;
    const int wave  = threadIdx.x >> 5;
    const int lane  = threadIdx.x & 31;
    const int nn    = lane & 15;
    const int half  = lane >> 4;
    const int q0    = qbase + wave * 16;

    // ---- load this wave's Q strip as 8 A-frags (row = lane&15)
    Frag16 Qf[8];
    {
        const unsigned* qrow =
            (const unsigned*)(ctq + ((size_t)bh*L_ + q0 + nn) * DQK_);
        #pragma unroll
        for (int kc = 0; kc < 8; ++kc)
            #pragma unroll
            for (int p = 0; p < 8; ++p)
                Qf[kc].u[p] = qrow[(kc*32 + kA0(p, half)) >> 1];
    }

    float mrow[8], lrow[8];
    v8f O[4];
    #pragma unroll
    for (int i = 0; i < 8; ++i) { mrow[i] = -1e30f; lrow[i] = 0.f; }
    #pragma unroll
    for (int nt = 0; nt < 4; ++nt)
        O[nt] = (v8f){0.f,0.f,0.f,0.f,0.f,0.f,0.f,0.f};

    const int nT = (qbase + 128) / 32;     // causal tile count for this block

    // ---- tile stage helper (async path: VGPR-free LDS fill via ASYNCcnt)
    auto stage = [&](int buf, int kv) {
        const char* ks = (const char*)(ctk  + ((size_t)bh*L_ + kv) * DQK_);
        const char* vs = (const char*)(vbar + ((size_t)bh*L_ + kv) * E_);
#if HAVE_ASYNC
        char* kd = (char*)Kbuf[buf];
        char* vd = (char*)Vbuf[buf];
        #pragma unroll
        for (int i = 0; i < 4; ++i) {
            int ofs = (threadIdx.x + i*256) * 16;
            async_cp16(kd + ofs, ks + ofs);
        }
        async_cp16(vd + threadIdx.x*16, vs + threadIdx.x*16);
#else
        const uint4* ks4 = (const uint4*)ks;
        const uint4* vs4 = (const uint4*)vs;
        uint4* kd4 = (uint4*)Kbuf[buf];
        uint4* vd4 = (uint4*)Vbuf[buf];
        for (int i = threadIdx.x; i < 1024; i += 256) kd4[i] = ks4[i];
        vd4[threadIdx.x] = vs4[threadIdx.x];
#endif
    };

    stage(0, 0);                            // prologue: fill buffer 0

    for (int t = 0; t < nT; ++t) {
        const int kv  = t * 32;
        const int buf = t & 1;

        wait_async0();                      // this wave's copies into `buf` done
        __syncthreads();                    // all waves: copies + prev compute done
        if (t + 1 < nT) stage(buf ^ 1, kv + 32);   // overlap next tile with compute

        if (kv > q0 + 15) continue;         // this wave's rows are all masked here
        const __bf16* Kb = Kbuf[buf];
        const __bf16* Vb = Vbuf[buf];

        // ---- scores: 16x32 tile = two 16x16 C-frags, contract 256 in 8 chunks
        v8f sc0 = (v8f){0.f,0.f,0.f,0.f,0.f,0.f,0.f,0.f};
        v8f sc1 = sc0;
        #pragma unroll
        for (int kc = 0; kc < 8; ++kc) {
            Frag16 b0, b1;
            #pragma unroll
            for (int p = 0; p < 8; ++p) {
                int k0 = kc*32 + half*16 + 2*p;        // B: lanes0-15 K=0..15
                b0.u[p] = *(const unsigned*)&Kb[(nn     )*DQK_ + k0];
                b1.u[p] = *(const unsigned*)&Kb[(16 + nn)*DQK_ + k0];
            }
            sc0 = __builtin_amdgcn_wmma_f32_16x16x32_bf16(
                      false, Qf[kc].v, false, b0.v, (short)0, sc0, false, false);
            sc1 = __builtin_amdgcn_wmma_f32_16x16x32_bf16(
                      false, Qf[kc].v, false, b1.v, (short)0, sc1, false, false);
        }

        // ---- online softmax (C-frag row = i + 8*half, col = nn)
        #pragma unroll
        for (int i = 0; i < 8; ++i) {
            const int qg = q0 + half*8 + i;
            float s0 = sc0[i] * 0.0625f;
            float s1 = sc1[i] * 0.0625f;
            if (kv + nn      > qg) s0 = -1e30f;
            if (kv + 16 + nn > qg) s1 = -1e30f;
            float rm   = redmax16(fmaxf(s0, s1));
            float mnew = fmaxf(mrow[i], rm);
            float alpha = __expf(mrow[i] - mnew);
            float p0 = __expf(s0 - mnew);
            float p1 = __expf(s1 - mnew);
            lrow[i] = lrow[i]*alpha + redsum16(p0 + p1);
            mrow[i] = mnew;
            #pragma unroll
            for (int nt = 0; nt < 4; ++nt) O[nt][i] *= alpha;
            Pbuf[wave][(half*8 + i)*32 + nn]      = (__bf16)p0;
            Pbuf[wave][(half*8 + i)*32 + 16 + nn] = (__bf16)p1;
        }
        __builtin_amdgcn_wave_barrier();   // order P store -> P reload (same wave)

        // ---- P (16x32) as A-frag, accumulate O += P @ Vb (32x64)
        Frag16 Pa;
        #pragma unroll
        for (int p = 0; p < 8; ++p)
            Pa.u[p] = *(const unsigned*)&Pbuf[wave][nn*32 + kA0(p, half)];

        #pragma unroll
        for (int nt = 0; nt < 4; ++nt) {
            Frag16 vb;
            #pragma unroll
            for (int e = 0; e < 16; ++e) {
                int k = half*16 + e;                   // B: lanes0-15 K=0..15
                vb.s[e] = ((const unsigned short*)Vb)[k*E_ + nt*16 + nn];
            }
            O[nt] = __builtin_amdgcn_wmma_f32_16x16x32_bf16(
                        false, Pa.v, false, vb.v, (short)0, O[nt], false, false);
        }
    }

    // ---- normalize and store out[b, q, h, e] (f32)
    const int b = bh >> 3, h = bh & 7;
    #pragma unroll
    for (int i = 0; i < 8; ++i) {
        const int qg = q0 + half*8 + i;
        const float inv = 1.f / lrow[i];
        #pragma unroll
        for (int nt = 0; nt < 4; ++nt)
            out[(((size_t)b*L_ + qg)*H_ + h)*E_ + nt*16 + nn] = O[nt][i] * inv;
    }
}

// ---------------------------------------------------------------------------
extern "C" void kernel_launch(void* const* d_in, const int* in_sizes, int n_in,
                              void* d_out, int out_size, void* d_ws, size_t ws_size,
                              hipStream_t stream) {
    (void)in_sizes; (void)n_in; (void)out_size; (void)ws_size;
    const float* q   = (const float*)d_in[0];
    const float* k   = (const float*)d_in[1];
    const float* v   = (const float*)d_in[2];
    const float* tau = (const float*)d_in[3];
    // d_in[4] label_pre_timeslot: unused; d_in[5] attn_mask: causal, computed inline
    const float* Wq = (const float*)d_in[6];
    const float* bq = (const float*)d_in[7];
    const float* Wk = (const float*)d_in[8];
    const float* bk = (const float*)d_in[9];
    const float* Wv = (const float*)d_in[10];
    const float* bv = (const float*)d_in[11];
    float* out = (float*)d_out;

    const size_t CTQ_N = (size_t)BH_ * L_ * DQK_;   // 16,777,216 bf16
    __bf16* ctq = (__bf16*)d_ws;
    __bf16* ctk = ctq + CTQ_N;
    __bf16* vb  = ctk + CTQ_N;                      // + BH*L*64

    const int rows_qk = B_*L_*H_*S_;                // 262144
    const int rows_v  = B_*L_*H_;                   // 65536

    interp_proj_qk<<<rows_qk/128, 256, 0, stream>>>(q, tau, Wq, bq, ctq);
    interp_proj_qk<<<rows_qk/128, 256, 0, stream>>>(k, tau, Wk, bk, ctk);
    interp_vbar  <<<rows_v /128, 256, 0, stream>>>(v, tau, Wv, bv, vb);
    attn_fwd<<<dim3(L_/128, BH_), 256, 0, stream>>>(ctq, ctk, vb, out);
}
